// Dempster_Shafer_module_40819369181539
// MI455X (gfx1250) — matrix operand order repaired and checked by reading, not compiled, not run on hardware
//
#include <hip/hip_runtime.h>
#include <math.h>

// Problem constants (from the reference): B=16384, F=512, P=128, C=10
#define B_N 16384
#define F_N 512
#define P_N 128
#define C_N 10
#define MBLK 64        // rows per block in main kernel
#define SI_STRIDE 129  // padded LDS row stride (129 % 64 == 1 -> conflict-free column walks)

typedef __attribute__((ext_vector_type(16))) __bf16 v16bf;
typedef __attribute__((ext_vector_type(8)))  float  v8f;
typedef __attribute__((ext_vector_type(16))) unsigned short us16v;

// cross-lane add of lane r and lane r^16 via one DS swizzle (SWAPX16)
__device__ __forceinline__ float xor16_add(float v) {
    int o = __builtin_amdgcn_ds_swizzle(__builtin_bit_cast(int, v), 0x401F);
    return v + __builtin_bit_cast(float, o);
}

// ---------------------------------------------------------------------------
// Prep kernel: one block per partition p.
//  - convert w[p][:] to bf16 (row-major, WMMA-B friendly)
//  - wnorm2[p] = ||w_p||^2  (f32 exact)
//  - gamma[p] = eta[p]^2, alpha[p] = sigmoid(xi[p])
//  - u[c][p] = beta[c][p]^2 / sum_c beta[c][p]^2
// ---------------------------------------------------------------------------
__global__ __launch_bounds__(256) void ds_prep(
    const float* __restrict__ w, const float* __restrict__ eta,
    const float* __restrict__ xi, const float* __restrict__ beta,
    unsigned short* __restrict__ wbf, float* __restrict__ wn2,
    float* __restrict__ gam, float* __restrict__ alp,
    float* __restrict__ u) {
    __shared__ float red[256];
    const int p = blockIdx.x;
    const int t = threadIdx.x;
    float s = 0.f;
    for (int k = t; k < F_N; k += 256) {
        float f = w[p * F_N + k];
        s = fmaf(f, f, s);
        wbf[p * F_N + k] = __builtin_bit_cast(unsigned short, (__bf16)f);
    }
    red[t] = s;
    __syncthreads();
    for (int off = 128; off > 0; off >>= 1) {
        if (t < off) red[t] += red[t + off];
        __syncthreads();
    }
    if (t == 0) {
        wn2[p] = red[0];
        float e = eta[p];
        gam[p] = e * e;
        alp[p] = 1.f / (1.f + __expf(-xi[p]));
        float b2[C_N];
        float denom = 0.f;
        #pragma unroll
        for (int c = 0; c < C_N; ++c) {
            float b = beta[c * P_N + p];
            b2[c] = b * b;
            denom += b2[c];
        }
        float inv = 1.f / denom;
        #pragma unroll
        for (int c = 0; c < C_N; ++c) u[c * P_N + p] = b2[c] * inv;
    }
}

// load + convert one A fragment (16x32 bf16) for this lane, accumulating ||x||^2
__device__ __forceinline__ v16bf load_a_frag(const float* __restrict__ base,
                                             float& nacc) {
    float4 f0 = ((const float4*)base)[0];
    float4 f1 = ((const float4*)base)[1];
    float4 f2 = ((const float4*)(base + 8))[0];
    float4 f3 = ((const float4*)(base + 8))[1];
    float fa[16] = {f0.x, f0.y, f0.z, f0.w, f1.x, f1.y, f1.z, f1.w,
                    f2.x, f2.y, f2.z, f2.w, f3.x, f3.y, f3.z, f3.w};
    v16bf r;
    #pragma unroll
    for (int j = 0; j < 16; ++j) {
        nacc = fmaf(fa[j], fa[j], nacc);
        r[j] = (__bf16)fa[j];   // native RNE conversion -> v_cvt_pk_bf16_f32
    }
    return r;
}

// ---------------------------------------------------------------------------
// Main kernel: 64 rows per block, 64 threads (2 waves), 32 rows (2 strips) per
// wave so each B fragment feeds two WMMAs.
// Phase 1: bf16 WMMA GEMM  acc = x_tile @ w^T  (+ ||x||^2 on the fly, f32 exact)
// Phase 2: si = alpha*exp(-gamma*d) -> LDS
// Phase 3: per-row sequential Dempster scan over p (non-associative -> in order)
// ---------------------------------------------------------------------------
__global__ __launch_bounds__(64) void ds_main(
    const float* __restrict__ x, const unsigned short* __restrict__ wbf,
    const float* __restrict__ wn2, const float* __restrict__ gam,
    const float* __restrict__ alp, const float* __restrict__ u,
    float* __restrict__ out) {

    __shared__ float si_lds[MBLK][SI_STRIDE];
    __shared__ float xnorm_lds[MBLK];
    __shared__ float u_lds[C_N][P_N];

    const int tid  = threadIdx.x;
    const int wave = tid >> 5;
    const int lane = tid & 31;
    const int half = lane >> 4;   // which 16-lane half of the wave
    const int r    = lane & 15;

    // stage u[c][p] into LDS (broadcast reads during the scan phase)
    for (int i = tid; i < C_N * P_N; i += 64)
        ((float*)u_lds)[i] = u[i];

    const int strip_base = wave * 32;                       // block-local row base
    const int arow0 = blockIdx.x * MBLK + strip_base + r;   // strip 0 row
    // A-fragment k-offsets (contiguous, matching B):
    // lanes 0-15 : elems 0..15 = K0+0..15
    // lanes 16-31: elems 0..15 = K0+16..31
    const float* a0base = x + (size_t)arow0 * F_N + half * 16;
    const float* a1base = a0base + (size_t)16 * F_N;        // strip 1 row (= +16 rows)

    v8f zero;
    #pragma unroll
    for (int j = 0; j < 8; ++j) zero[j] = 0.f;
    v8f acc[2][8];
    #pragma unroll
    for (int s = 0; s < 2; ++s)
        #pragma unroll
        for (int nt = 0; nt < 8; ++nt) acc[s][nt] = zero;

    float norm0 = 0.f, norm1 = 0.f;

    for (int kk = 0; kk < F_N / 32; ++kk) {
        const int K0 = kk * 32;
        v16bf av0 = load_a_frag(a0base + K0, norm0);
        v16bf av1 = load_a_frag(a1base + K0, norm1);

        // B fragment (32x16 bf16): column p = nt*16 + r,
        // lanes 0-15 hold K0+0..15, lanes 16-31 hold K0+16..31 (contiguous)
        const unsigned short* bbase = wbf + K0 + half * 16;
        #pragma unroll
        for (int nt = 0; nt < 8; ++nt) {
            const int p = nt * 16 + r;
            us16v ubv = *(const us16v*)(bbase + (size_t)p * F_N);
            v16bf bv = __builtin_bit_cast(v16bf, ubv);
            acc[0][nt] = __builtin_amdgcn_wmma_f32_16x16x32_bf16(
                false, av0, false, bv, (short)0, acc[0][nt], false, false);
            acc[1][nt] = __builtin_amdgcn_wmma_f32_16x16x32_bf16(
                false, av1, false, bv, (short)0, acc[1][nt], false, false);
        }
    }

    // finish ||x_row||^2: lane r and lane r+16 hold the two K-halves of a row
    float rn0 = xor16_add(norm0);
    float rn1 = xor16_add(norm1);
    if (half == 0) {
        xnorm_lds[strip_base + r]      = rn0;
        xnorm_lds[strip_base + 16 + r] = rn1;
    }
    __syncthreads();

    // ---- si = alpha * exp(-gamma * d), d = ||x||^2 + ||w||^2 - 2 x.w
    // C-tile layout: lanes 0-15 -> rows i (vgpr i), lanes 16-31 -> rows i+8
    #pragma unroll
    for (int nt = 0; nt < 8; ++nt) {
        const int p = nt * 16 + r;
        const float wn = wn2[p];
        const float g  = gam[p];
        const float al = alp[p];
        #pragma unroll
        for (int s = 0; s < 2; ++s) {
            #pragma unroll
            for (int i = 0; i < 8; ++i) {
                const int row_local = strip_base + s * 16 + (half ? i + 8 : i);
                float d = xnorm_lds[row_local] + wn - 2.0f * acc[s][nt][i];
                si_lds[row_local][p] = al * __expf(-g * d);
            }
        }
    }
    __syncthreads();

    // ---- sequential Dempster combination scan (one thread per row; the step
    // operation is NOT associative, so it must run in reference order)
    {
        const int row_local = tid;   // blockDim.x == MBLK
        float mx = -1e30f;
        for (int p = 0; p < P_N; ++p) mx = fmaxf(mx, si_lds[row_local][p]);
        const float inv = 1.0f / (mx + 1e-4f);

        float m1[C_N + 1];
        {
            float s = si_lds[row_local][0] * inv;
            #pragma unroll
            for (int c = 0; c < C_N; ++c) m1[c] = u_lds[c][0] * s;
            m1[C_N] = 1.0f - s;
        }
        for (int p = 1; p < P_N; ++p) {
            float s = si_lds[row_local][p] * inv;
            float m2[C_N + 1];
            #pragma unroll
            for (int c = 0; c < C_N; ++c) m2[c] = u_lds[c][p] * s;
            m2[C_N] = 1.0f - s;
            const float o1 = m1[C_N], o2 = m2[C_N];
            float cmb[C_N + 1];
            float sum = 0.f;
            #pragma unroll
            for (int j = 0; j <= C_N; ++j) {
                cmb[j] = m1[j] * m2[j] + m1[j] * o2 + o1 * m2[j];
                sum += cmb[j];
            }
            const float isum = 1.0f / sum;
            #pragma unroll
            for (int j = 0; j <= C_N; ++j) m1[j] = cmb[j] * isum;
        }
        float sum = 0.f;
        #pragma unroll
        for (int j = 0; j <= C_N; ++j) sum += m1[j];
        const float isum = 1.0f / sum;
        const size_t row = (size_t)blockIdx.x * MBLK + row_local;
        #pragma unroll
        for (int j = 0; j <= C_N; ++j) out[row * (C_N + 1) + j] = m1[j] * isum;
    }
}

// ---------------------------------------------------------------------------
extern "C" void kernel_launch(void* const* d_in, const int* in_sizes, int n_in,
                              void* d_out, int out_size, void* d_ws, size_t ws_size,
                              hipStream_t stream) {
    (void)in_sizes; (void)n_in; (void)out_size; (void)ws_size;
    const float* x    = (const float*)d_in[0];
    const float* w    = (const float*)d_in[1];
    const float* eta  = (const float*)d_in[2];
    const float* xi   = (const float*)d_in[3];
    const float* beta = (const float*)d_in[4];
    float* out = (float*)d_out;

    char* ws = (char*)d_ws;
    unsigned short* wbf = (unsigned short*)ws;            // 128*512*2 = 131072 B
    float* wn2 = (float*)(ws + 131072);                   // 512 B
    float* gam = (float*)(ws + 131072 + 512);             // 512 B
    float* alp = (float*)(ws + 131072 + 1024);            // 512 B
    float* u   = (float*)(ws + 131072 + 1536);            // 10*128*4 = 5120 B

    ds_prep<<<P_N, 256, 0, stream>>>(w, eta, xi, beta, wbf, wn2, gam, alp, u);
    ds_main<<<B_N / MBLK, 64, 0, stream>>>(x, wbf, wn2, gam, alp, u, out);
}
